// SwitchFeedForward_9929964389239
// MI455X (gfx1250) — matrix operand order repaired and checked by reading, not compiled, not run on hardware
//
#include <hip/hip_runtime.h>
#include <hip/hip_bf16.h>
#include <stdint.h>

typedef __attribute__((ext_vector_type(16))) __bf16 v16bf;
typedef __attribute__((ext_vector_type(8)))  float  v8f;
typedef int v4i_vs __attribute__((vector_size(16)));   // matches builtin param

__device__ __forceinline__ uint16_t f32_to_bf16_bits(float f) {
  union { float f; uint32_t u; } v; v.f = f;
  uint32_t r = v.u + 0x7FFFu + ((v.u >> 16) & 1u);   // round-to-nearest-even
  return (uint16_t)(r >> 16);
}

// ---- CDNA5 async global->LDS copy (ASYNCcnt) with safe fallback -------------
// AS(1) global ptr == generic bits (identity map); AS(3) ptr == low 32 bits of
// the generic LDS address.
__device__ __forceinline__ void async_copy_b128(const void* g, void* l) {
#if __has_builtin(__builtin_amdgcn_global_load_async_to_lds_b128)
  __builtin_amdgcn_global_load_async_to_lds_b128(
      (__attribute__((address_space(1))) v4i_vs*)(uintptr_t)g,
      (__attribute__((address_space(3))) v4i_vs*)(uint32_t)(uintptr_t)l,
      0, 0);
#else
  *reinterpret_cast<uint4*>(l) = *reinterpret_cast<const uint4*>(g);
#endif
}

__device__ __forceinline__ void wait_async0() {
#if __has_builtin(__builtin_amdgcn_s_wait_asynccnt)
  __builtin_amdgcn_s_wait_asynccnt(0);
#else
  asm volatile("s_wait_asynccnt 0x0" ::: "memory");
#endif
}

// ---------------- Router: logits -> softmax top-1 (prob_max, route) ----------
__global__ __launch_bounds__(256)
void router_kernel(const float* __restrict__ x, const float* __restrict__ sw,
                   const float* __restrict__ sb, float* __restrict__ prob,
                   int* __restrict__ route, int N) {
  const int D = 768, E = 8;
  int wv = (blockIdx.x * blockDim.x + threadIdx.x) >> 5;  // one wave per token
  int lane = threadIdx.x & 31;
  if (wv >= N) return;
  const float* xr = x + (size_t)wv * D;
  float acc[E];
#pragma unroll
  for (int e = 0; e < E; ++e) acc[e] = 0.f;
  for (int d = lane; d < D; d += 32) {
    float xv = xr[d];
    const float* wr = sw + (size_t)d * E;
#pragma unroll
    for (int e = 0; e < E; ++e) acc[e] += xv * wr[e];
  }
#pragma unroll
  for (int e = 0; e < E; ++e) {
#pragma unroll
    for (int off = 16; off >= 1; off >>= 1)
      acc[e] += __shfl_xor(acc[e], off, 32);
  }
  if (lane == 0) {
    float lmax = -3.4e38f; int arg = 0;
#pragma unroll
    for (int e = 0; e < E; ++e) {
      float l = acc[e] + sb[e];
      acc[e] = l;
      if (l > lmax) { lmax = l; arg = e; }   // strict '>' == argmax-first tie rule
    }
    float s = 0.f;
#pragma unroll
    for (int e = 0; e < E; ++e) s += expf(acc[e] - lmax);
    prob[wv] = 1.f / s;
    route[wv] = arg;
  }
}

// ---------------- Rank within expert by (prob desc, index asc) ---------------
__global__ __launch_bounds__(256)
void rank_kernel(const float* __restrict__ prob, const int* __restrict__ route,
                 int* __restrict__ rank, int N) {
  __shared__ float sp[2048];
  __shared__ int   sr[2048];
  int wv = threadIdx.x >> 5;
  int lane = threadIdx.x & 31;
  int token = blockIdx.x * 8 + wv;
  float pi = prob[token];
  int   ri = route[token];
  int cnt = 0;
  for (int base = 0; base < N; base += 2048) {
    __syncthreads();
    for (int j = threadIdx.x; j < 2048; j += 256) {
      sp[j] = prob[base + j];
      sr[j] = route[base + j];
    }
    __syncthreads();
    for (int j = lane; j < 2048; j += 32) {
      int jg = base + j;
      bool c = (sr[j] == ri) && ((sp[j] > pi) || ((sp[j] == pi) && (jg < token)));
      cnt += c ? 1 : 0;
    }
  }
#pragma unroll
  for (int off = 16; off >= 1; off >>= 1) cnt += __shfl_xor(cnt, off, 32);
  if (lane == 0) rank[token] = cnt;
}

// ---------------- Dispatch kept tokens into bf16 buf[E][cap][D] --------------
__global__ __launch_bounds__(256)
void dispatch_kernel(const float* __restrict__ x, const int* __restrict__ route,
                     const int* __restrict__ rank, uint16_t* __restrict__ buf, int cap) {
  const int D = 768;
  int t = blockIdx.x;
  int rk = rank[t];
  if (rk >= cap) return;
  int e = route[t];
  const float* xr = x + (size_t)t * D;
  uint16_t* dst = buf + ((size_t)e * cap + rk) * D;
  for (int d = threadIdx.x; d < D; d += 256) dst[d] = f32_to_bf16_bits(xr[d]);
}

// -------- Weight prep: W[e][K][N] f32 -> WT[e][N][K] bf16 (transpose) --------
__global__ __launch_bounds__(256)
void transpose_bf16_kernel(const float* __restrict__ W, uint16_t* __restrict__ WT,
                           int K, int N) {
  __shared__ float tile[32][33];
  int e = blockIdx.z;
  int k0 = blockIdx.y * 32, n0 = blockIdx.x * 32;
  int tx = threadIdx.x & 31, ty = threadIdx.x >> 5;   // 32 x 8
  const float* src = W + (size_t)e * K * N;
#pragma unroll
  for (int r = ty; r < 32; r += 8)
    tile[r][tx] = src[(size_t)(k0 + r) * N + n0 + tx];
  __syncthreads();
  uint16_t* dst = WT + (size_t)e * (size_t)K * N;
#pragma unroll
  for (int r = ty; r < 32; r += 8)
    dst[(size_t)(n0 + r) * K + k0 + tx] = f32_to_bf16_bits(tile[tx][r]);
}

// ---------------- Grouped GEMM: C = A(bf16) x BT(bf16) + bias ----------------
// A: [E][M][K] bf16 row-major.  BT: [E][N][K] bf16 (K contiguous).
// Double-buffered LDS stages filled with async global->LDS B128 copies.
// GELU_BF16_OUT=1: fused exact-erf GELU, bf16 out (h). =0: bias only, f32 out.
#define BM 128
#define BN 256
#define BK 32
#define LDP 40   // padded row: 80B -> 16B-aligned chunks, conflict-free frags

template<int GELU_BF16_OUT>
__global__ __launch_bounds__(256)
void moe_gemm_kernel(const uint16_t* __restrict__ A,
                     const uint16_t* __restrict__ BT,
                     const float* __restrict__ bias,
                     void* __restrict__ C,
                     int M, int K, int N) {
  __shared__ __align__(16) uint16_t As[2][BM * LDP];  // 2 x 10240 B
  __shared__ __align__(16) uint16_t Bs[2][BN * LDP];  // 2 x 20480 B

  const int e  = blockIdx.z;
  const int m0 = blockIdx.y * BM;
  const int n0 = blockIdx.x * BN;

  const uint16_t* gA  = A  + ((size_t)e * M + m0) * K;
  const uint16_t* gBT = BT + ((size_t)e * N + n0) * K;

  const int tid  = threadIdx.x;
  const int wv   = tid >> 5;
  const int lane = tid & 31;
  const int wm   = wv & 1;              // 2 waves along M (64 rows each)
  const int wn   = wv >> 1;             // 4 waves along N (64 cols each)
  const int lm   = lane & 15;
  const int lh   = lane >> 4;

  auto stage = [&](int ks, int bufi) {
    const uint16_t* gAk = gA  + ks * BK;
    const uint16_t* gBk = gBT + ks * BK;
#pragma unroll
    for (int j = 0; j < 2; ++j) {       // A: 512 chunks of 8 bf16 (16B)
      int c = tid + 256 * j;
      int row = c >> 2, col = (c & 3) * 8;
      async_copy_b128(gAk + (size_t)row * K + col, &As[bufi][row * LDP + col]);
    }
#pragma unroll
    for (int j = 0; j < 4; ++j) {       // B: 1024 chunks of 8 bf16 (16B)
      int c = tid + 256 * j;
      int row = c >> 2, col = (c & 3) * 8;
      async_copy_b128(gBk + (size_t)row * K + col, &Bs[bufi][row * LDP + col]);
    }
  };

  v8f acc[4][4];
#pragma unroll
  for (int i = 0; i < 4; ++i)
#pragma unroll
    for (int j = 0; j < 4; ++j) acc[i][j] = 0.0f;

  const int nk = K / BK;
  stage(0, 0);
  wait_async0();
  __syncthreads();

  for (int ks = 0; ks < nk; ++ks) {
    int cur = ks & 1;
    if (ks + 1 < nk) stage(ks + 1, cur ^ 1);   // overlap copies with WMMA

    union Frag { uint32_t u[8]; v16bf v; };
    Frag fa[4], fb[4];
    const uint16_t* Ab = As[cur];
    const uint16_t* Bb = Bs[cur];
    // A frag: lanes 0-15 K=0..15, lanes 16-31 K=16..31 (matches B)
#pragma unroll
    for (int mi = 0; mi < 4; ++mi) {
      int row = wm * 64 + mi * 16 + lm;
#pragma unroll
      for (int i = 0; i < 8; ++i) {
        int kk = lh * 16 + 2 * i;
        fa[mi].u[i] = *reinterpret_cast<const uint32_t*>(&Ab[row * LDP + kk]);
      }
    }
    // B frag: lanes 0-15 K=0..15, lanes 16-31 K=16..31
#pragma unroll
    for (int ni = 0; ni < 4; ++ni) {
      int row = wn * 64 + ni * 16 + lm;
#pragma unroll
      for (int i = 0; i < 8; ++i) {
        int kk = lh * 16 + 2 * i;
        fb[ni].u[i] = *reinterpret_cast<const uint32_t*>(&Bb[row * LDP + kk]);
      }
    }
#pragma unroll
    for (int mi = 0; mi < 4; ++mi)
#pragma unroll
      for (int ni = 0; ni < 4; ++ni)
        acc[mi][ni] = __builtin_amdgcn_wmma_f32_16x16x32_bf16(
            false, fa[mi].v, false, fb[ni].v, (short)0, acc[mi][ni], false, false);

    wait_async0();       // own async copies for next stage complete
    __syncthreads();     // all waves done copying + done reading this buffer
  }

  // epilogue: C/D layout M = r + 8*lh, N = lm
#pragma unroll
  for (int ni = 0; ni < 4; ++ni) {
    int col = n0 + wn * 64 + ni * 16 + lm;
    float bv = bias[(size_t)e * N + col];
#pragma unroll
    for (int mi = 0; mi < 4; ++mi) {
#pragma unroll
      for (int r = 0; r < 8; ++r) {
        int row = m0 + wm * 64 + mi * 16 + r + 8 * lh;
        float v = acc[mi][ni][r] + bv;
        size_t idx = ((size_t)e * M + row) * N + col;
        if (GELU_BF16_OUT) {
          v = 0.5f * v * (1.0f + erff(v * 0.70710678118654752f));  // exact GELU
          reinterpret_cast<uint16_t*>(C)[idx] = f32_to_bf16_bits(v);
        } else {
          reinterpret_cast<float*>(C)[idx] = v;
        }
      }
    }
  }
}

// ---------------- Combine: kept -> expert out, dropped -> passthrough --------
__global__ __launch_bounds__(256)
void combine_kernel(const float* __restrict__ x, const float* __restrict__ o,
                    const float* __restrict__ prob, const int* __restrict__ route,
                    const int* __restrict__ rank, float* __restrict__ out, int cap) {
  const int D = 768;
  int t = blockIdx.x;
  int rk = rank[t];
  float p = prob[t];
  const float* src = (rk < cap)
      ? (o + (((size_t)route[t] * cap + rk) * D))
      : (x + (size_t)t * D);
  float* dst = out + (size_t)t * D;
  for (int d = threadIdx.x; d < D; d += 256) dst[d] = src[d] * p;
}

extern "C" void kernel_launch(void* const* d_in, const int* in_sizes, int n_in,
                              void* d_out, int out_size, void* d_ws, size_t ws_size,
                              hipStream_t stream) {
  const float* x  = (const float*)d_in[0];  // [8,2048,768]
  const float* sw = (const float*)d_in[1];  // [768,8]
  const float* sb = (const float*)d_in[2];  // [8]
  const float* w1 = (const float*)d_in[3];  // [8,768,3072]
  const float* b1 = (const float*)d_in[4];  // [8,3072]
  const float* w2 = (const float*)d_in[5];  // [8,3072,768]
  const float* b2 = (const float*)d_in[6];  // [8,768]
  float* out = (float*)d_out;

  const int N = 16384, D = 768, F = 3072, E = 8;
  const int cap = 2560;                      // int(1.25 * N / E)

  char* p = (char*)d_ws;
  auto alloc = [&](size_t bytes) {
    char* r = p; p += (bytes + 255) & ~(size_t)255; return r;
  };
  float*    prob  = (float*)alloc((size_t)N * 4);
  int*      route = (int*)alloc((size_t)N * 4);
  int*      rank  = (int*)alloc((size_t)N * 4);
  uint16_t* buf   = (uint16_t*)alloc((size_t)E * cap * D * 2);   // 31.5 MB
  uint16_t* h     = (uint16_t*)alloc((size_t)E * cap * F * 2);   // 126 MB
  float*    o     = (float*)alloc((size_t)E * cap * D * 4);      // 63 MB
  uint16_t* w1T   = (uint16_t*)alloc((size_t)E * D * F * 2);     // 37.7 MB [E][F][D]
  uint16_t* w2T   = (uint16_t*)alloc((size_t)E * F * D * 2);     // 37.7 MB [E][D][F]

  (void)hipMemsetAsync(buf, 0, (size_t)E * cap * D * 2, stream); // empty slots = 0

  router_kernel  <<<N / 8, 256, 0, stream>>>(x, sw, sb, prob, route, N);
  rank_kernel    <<<N / 8, 256, 0, stream>>>(prob, route, rank, N);
  dispatch_kernel<<<N,     256, 0, stream>>>(x, route, rank, buf, cap);
  // w1 [D][F] -> w1T [F][D] ; w2 [F][D] -> w2T [D][F]  (bf16)
  transpose_bf16_kernel<<<dim3(F / 32, D / 32, E), 256, 0, stream>>>(w1, w1T, D, F);
  transpose_bf16_kernel<<<dim3(D / 32, F / 32, E), 256, 0, stream>>>(w2, w2T, F, D);
  moe_gemm_kernel<1><<<dim3(F / BN, cap / BM, E), 256, 0, stream>>>(buf, w1T, b1, h, cap, D, F);
  moe_gemm_kernel<0><<<dim3(D / BN, cap / BM, E), 256, 0, stream>>>(h,   w2T, b2, o, cap, F, D);
  combine_kernel <<<N,     256, 0, stream>>>(x, o, prob, route, rank, out, cap);
}
